// QLoRALinear_9199819948155
// MI455X (gfx1250) — compile-verified, matrix operand order checked by reading, not run on hardware
//
#include <hip/hip_runtime.h>
#include <hip/hip_bf16.h>

// ---------------------------------------------------------------------------
// QLoRA linear for MI455X (gfx1250, wave32):
//   out = x @ dequant_nf4(codes, absmax)^T + bias + 2.0 * (x @ A) @ B
// Pipeline:
//   k_dequant : NF4 codes -> bf16 W [N,K]   (one pass; W bf16 = 33.5MB, L2-resident)
//   k_xconv   : x fp32 -> bf16 [M,K] and fused t = x @ lora_A  [M,8]
//   k_gemm    : bf16 WMMA GEMM, 128x128 block tile, 4 waves x (64x64) wave tiles,
//               double-buffered LDS fed by global_load_async_to_lds_b128 (ASYNCcnt),
//               f32 accum, fused bias+LoRA epilogue
// ---------------------------------------------------------------------------

typedef __attribute__((ext_vector_type(16))) __bf16 v16bf;
typedef __attribute__((ext_vector_type(8)))  float  v8f;

union FragBF { v16bf v; uint4 q[2]; };

__device__ __constant__ float c_nf4[16] = {
    -1.0f, -0.6961928009986877f, -0.5250730514526367f, -0.39491748809814453f,
    -0.28444138169288635f, -0.18477343022823334f, -0.09105003625154495f, 0.0f,
    0.07958029955625534f, 0.16093020141124725f, 0.24611230194568634f,
    0.33791524171829224f, 0.44070982933044434f, 0.5626170039176941f,
    0.7229568362236023f, 1.0f};

__device__ __forceinline__ unsigned short f2bf(float f) {
    union { float f; unsigned u; } v; v.f = f;
    unsigned r = v.u + 0x7FFFu + ((v.u >> 16) & 1u);   // round-to-nearest-even
    return (unsigned short)(r >> 16);
}

// -------------------------------- dequant ----------------------------------
__global__ void k_dequant(const int* __restrict__ codes,
                          const float* __restrict__ absmax,
                          unsigned short* __restrict__ wb,
                          int K, long total8) {
    __shared__ float lut[16];
    if (threadIdx.x < 16) lut[threadIdx.x] = c_nf4[threadIdx.x];
    __syncthreads();
    long tid = (long)blockIdx.x * blockDim.x + threadIdx.x;
    if (tid >= total8) return;
    size_t base = (size_t)tid * 8;
    int o  = (int)(base / (size_t)K);
    int ci = (int)(base % (size_t)K);
    float am = absmax[(size_t)o * (K >> 6) + (ci >> 6)];
    int4 c0 = *(const int4*)(codes + base);
    int4 c1 = *(const int4*)(codes + base + 4);
    union { unsigned short s[8]; uint4 q; } r;
    r.s[0] = f2bf(lut[c0.x] * am);
    r.s[1] = f2bf(lut[c0.y] * am);
    r.s[2] = f2bf(lut[c0.z] * am);
    r.s[3] = f2bf(lut[c0.w] * am);
    r.s[4] = f2bf(lut[c1.x] * am);
    r.s[5] = f2bf(lut[c1.y] * am);
    r.s[6] = f2bf(lut[c1.z] * am);
    r.s[7] = f2bf(lut[c1.w] * am);
    *(uint4*)(wb + base) = r.q;
}

// ----------------------- x -> bf16 + t = x @ lora_A ------------------------
__global__ void k_xconv(const float* __restrict__ x,
                        const float* __restrict__ A,     // [K,8]
                        unsigned short* __restrict__ xb, // [M,K] bf16
                        float* __restrict__ tq,          // [M,8]
                        int K) {
    const int row = blockIdx.x;
    const int tid = threadIdx.x;
    const size_t rb = (size_t)row * K;
    float s0=0,s1=0,s2=0,s3=0,s4=0,s5=0,s6=0,s7=0;
    for (int k = tid; k < K; k += 256) {
        float v = x[rb + k];
        xb[rb + k] = f2bf(v);
        float4 a0 = *(const float4*)(A + (size_t)k * 8);
        float4 a1 = *(const float4*)(A + (size_t)k * 8 + 4);
        s0 += v * a0.x; s1 += v * a0.y; s2 += v * a0.z; s3 += v * a0.w;
        s4 += v * a1.x; s5 += v * a1.y; s6 += v * a1.z; s7 += v * a1.w;
    }
    __shared__ float red[8 * 256];
    red[0*256+tid]=s0; red[1*256+tid]=s1; red[2*256+tid]=s2; red[3*256+tid]=s3;
    red[4*256+tid]=s4; red[5*256+tid]=s5; red[6*256+tid]=s6; red[7*256+tid]=s7;
    __syncthreads();
    for (int st = 128; st > 0; st >>= 1) {
        if (tid < st) {
            #pragma unroll
            for (int r = 0; r < 8; ++r)
                red[r*256+tid] += red[r*256+tid+st];
        }
        __syncthreads();
    }
    if (tid < 8) tq[(size_t)row * 8 + tid] = red[tid * 256];
}

// ------------------------------- WMMA GEMM ---------------------------------
#define TM  128
#define TN  128
#define KT  32
#define STR 40   // padded LDS row stride (bf16 elems): 80B rows -> conflict-free
#define NTHR 128 // 4 waves, arranged 2(M) x 2(N), each wave computes 64x64

__global__ __launch_bounds__(NTHR)
void k_gemm(const unsigned short* __restrict__ xb,   // [M,K] bf16
            const unsigned short* __restrict__ wb,   // [N,K] bf16
            const float* __restrict__ tq,            // [M,8]
            const float* __restrict__ loraB,         // [8,N]
            const float* __restrict__ bias,          // [N]
            float* __restrict__ out,                 // [M,N]
            int K, int N) {
    __shared__ unsigned short ldsX[2][TM * STR];     // 2 x 10240 B
    __shared__ unsigned short ldsW[2][TN * STR];     // 2 x 10240 B
    __shared__ float ldsT[TM * 8];
    __shared__ float ldsLB[8 * TN];
    __shared__ float ldsBias[TN];

    const int tid  = threadIdx.x;
    const int lane = tid & 31;
    const int w    = tid >> 5;          // 4 waves: 2 (M) x 2 (N)
    const int wm   = w & 1, wn = w >> 1;
    const int h    = lane >> 4, r16 = lane & 15;
    const int m0 = blockIdx.y * TM, n0 = blockIdx.x * TN;

    // stage LoRA epilogue operands once per workgroup
    for (int i = tid; i < TM * 8; i += NTHR) ldsT[i] = tq[(size_t)m0 * 8 + i];
    for (int i = tid; i < 8 * TN; i += NTHR)
        ldsLB[i] = loraB[(size_t)(i >> 7) * N + n0 + (i & 127)];
    for (int i = tid; i < TN; i += NTHR) ldsBias[i] = bias[n0 + i];

    // async staging: 512 chunks of 16B per 128x32 tile, 4 chunks per thread,
    // for X and W -> 8 global_load_async_to_lds_b128 per thread per K-step
    const int nk = K / KT;

    auto issue = [&](int buf, int kt) {
        const size_t kofs = (size_t)kt * KT;
        #pragma unroll
        for (int j = 0; j < 4; ++j) {
            const int c = j * NTHR + tid;     // chunk id 0..511
            const int row = c >> 2;           // 4 chunks (64B) per row
            const int col = (c & 3) * 8;      // element offset within K-tile
            const unsigned short* gp = xb + (size_t)(m0 + row) * K + kofs + col;
            unsigned la = (unsigned)(uintptr_t)&ldsX[buf][row * STR + col];
            asm volatile("global_load_async_to_lds_b128 %0, %1, off"
                         :: "v"(la), "v"(gp) : "memory");
        }
        #pragma unroll
        for (int j = 0; j < 4; ++j) {
            const int c = j * NTHR + tid;
            const int row = c >> 2;
            const int col = (c & 3) * 8;
            const unsigned short* gp = wb + (size_t)(n0 + row) * K + kofs + col;
            unsigned la = (unsigned)(uintptr_t)&ldsW[buf][row * STR + col];
            asm volatile("global_load_async_to_lds_b128 %0, %1, off"
                         :: "v"(la), "v"(gp) : "memory");
        }
    };

    const v8f vz = {0.f,0.f,0.f,0.f,0.f,0.f,0.f,0.f};
    v8f acc[4][4];
    #pragma unroll
    for (int i = 0; i < 4; ++i)
        #pragma unroll
        for (int j = 0; j < 4; ++j) acc[i][j] = vz;

    issue(0, 0);                        // prologue: fill buffer 0

    for (int kt = 0; kt < nk; ++kt) {
        const int cur = kt & 1;
        if (kt + 1 < nk) {
            issue(cur ^ 1, kt + 1);     // overlap next tile's async DMA
            asm volatile("s_wait_asynccnt 0x8" ::: "memory");
        } else {
            asm volatile("s_wait_asynccnt 0x0" ::: "memory");
        }
        __syncthreads();                // buf[cur] valid block-wide

        // fragment reads per ISA 16-bit A/B layout:
        // lane half h holds K in {8h..8h+7} (frag VGPR0-3) and {16+8h..} (VGPR4-7)
        FragBF a[4], b[4];
        #pragma unroll
        for (int fi = 0; fi < 4; ++fi) {
            const uint4* p = (const uint4*)&ldsX[cur][(wm*64 + fi*16 + r16) * STR + 8*h];
            a[fi].q[0] = p[0];          // +0  elems
            a[fi].q[1] = p[2];          // +16 elems
        }
        #pragma unroll
        for (int bi = 0; bi < 4; ++bi) {
            const uint4* p = (const uint4*)&ldsW[cur][(wn*64 + bi*16 + r16) * STR + 8*h];
            b[bi].q[0] = p[0];
            b[bi].q[1] = p[2];
        }
        #pragma unroll
        for (int fi = 0; fi < 4; ++fi)
            #pragma unroll
            for (int bi = 0; bi < 4; ++bi)
                acc[fi][bi] = __builtin_amdgcn_wmma_f32_16x16x32_bf16(
                    false, a[fi].v, false, b[bi].v,
                    (short)0, acc[fi][bi], false, false);

        __syncthreads();                // all waves done reading buf[cur]
    }

    // epilogue: out = acc + bias + 2.0 * (t @ loraB)
    // C/D layout: lane(0-15): N=lane, M=vgpr; lane(16-31): N=lane-16, M=8+vgpr
    #pragma unroll
    for (int bi = 0; bi < 4; ++bi) {
        const int nl = wn * 64 + bi * 16 + r16;
        const float bv = ldsBias[nl];
        float Bc[8];
        #pragma unroll
        for (int r = 0; r < 8; ++r) Bc[r] = ldsLB[r * TN + nl];
        #pragma unroll
        for (int fi = 0; fi < 4; ++fi) {
            const int mlb = wm * 64 + fi * 16 + 8 * h;
            float* op = out + (size_t)(m0 + mlb) * N + (n0 + nl);
            #pragma unroll
            for (int v = 0; v < 8; ++v) {
                const float* tp = &ldsT[(mlb + v) * 8];
                float lr = tp[0]*Bc[0] + tp[1]*Bc[1] + tp[2]*Bc[2] + tp[3]*Bc[3]
                         + tp[4]*Bc[4] + tp[5]*Bc[5] + tp[6]*Bc[6] + tp[7]*Bc[7];
                op[(size_t)v * N] = acc[fi][bi][v] + bv + 2.0f * lr;
            }
        }
    }
}

// ------------------------------- launcher ----------------------------------
extern "C" void kernel_launch(void* const* d_in, const int* in_sizes, int n_in,
                              void* d_out, int out_size, void* d_ws, size_t ws_size,
                              hipStream_t stream) {
    const float* x       = (const float*)d_in[0];
    const int*   w_codes = (const int*)  d_in[1];
    const float* absmax  = (const float*)d_in[2];
    const float* bias    = (const float*)d_in[3];
    const float* lora_A  = (const float*)d_in[4];
    const float* lora_B  = (const float*)d_in[5];
    float* out = (float*)d_out;

    const int  O = in_sizes[3];                 // 4096
    const int  I = in_sizes[1] / O;             // 4096 (K)
    const long M = (long)in_sizes[0] / I;       // 8192

    // workspace: xb bf16 [M,I] | wb bf16 [O,I] | t fp32 [M,8]
    unsigned short* xb = (unsigned short*)d_ws;
    unsigned short* wb = (unsigned short*)((char*)d_ws + (size_t)M * I * 2);
    float* tq = (float*)((char*)d_ws + (size_t)M * I * 2 + (size_t)O * I * 2);

    // 1) dequantize NF4 -> bf16 W
    long total8 = (long)O * I / 8;
    int blks = (int)((total8 + 255) / 256);
    k_dequant<<<blks, 256, 0, stream>>>(w_codes, absmax, wb, I, total8);

    // 2) x -> bf16, fused t = x @ lora_A
    k_xconv<<<(int)M, 256, 0, stream>>>(x, lora_A, xb, tq, I);

    // 3) main bf16 WMMA GEMM with fused bias + LoRA epilogue
    dim3 grid(O / TN, (int)(M / TM));
    k_gemm<<<grid, NTHR, 0, stream>>>(xb, wb, tq, lora_B, bias, out, I, O);
}
